// BnFireFc_69630009803372
// MI455X (gfx1250) — matrix operand order repaired and checked by reading, not compiled, not run on hardware
//
#include <hip/hip_runtime.h>
#include <hip/hip_bf16.h>

typedef __attribute__((ext_vector_type(16))) _Float16 v16h;
typedef __attribute__((ext_vector_type(8)))  float    v8f;
typedef unsigned int u32x4 __attribute__((ext_vector_type(4)));
typedef int          i32x8 __attribute__((ext_vector_type(8)));
typedef int          i32x4 __attribute__((ext_vector_type(4)));

#define CIN  512
#define COUT 512
#define EPS_BN 1e-5f

// ---------------- scratch layout (bytes) ----------------
// [0,2048)      : per-channel sum       (512 f32)
// [2048,4096)   : per-channel sumsq     (512 f32)
// [4096,6144)   : per-channel threshold (512 f32)
// [8192,532480) : packed W in f16 fragment order (512*512 halfs)

__global__ __launch_bounds__(256) void init_stats_kernel(float* ws) {
    int i = blockIdx.x * 256 + threadIdx.x;
    if (i < 1024) ws[i] = 0.0f;
}

// Column-wise sum / sumsq over x[N, CIN]; coalesced reads, f32 atomics.
__global__ __launch_bounds__(512) void bn_stats_kernel(const float* __restrict__ x,
                                                       float* __restrict__ sum,
                                                       float* __restrict__ sumsq,
                                                       int N) {
    int c = threadIdx.x;                 // 512 threads = one per channel
    float s = 0.0f, sq = 0.0f;
    for (int r = blockIdx.x; r < N; r += gridDim.x) {
        float v = x[(size_t)r * CIN + c];
        s += v;
        sq = fmaf(v, v, sq);
    }
    atomicAdd(&sum[c], s);
    atomicAdd(&sumsq[c], sq);
}

// t[c] = mean - bias_bn*sqrt(var+eps)/weight_bn   (valid since weight_bn > 0)
__global__ __launch_bounds__(512) void threshold_kernel(const float* __restrict__ sum,
                                                        const float* __restrict__ sumsq,
                                                        const float* __restrict__ w_bn,
                                                        const float* __restrict__ b_bn,
                                                        float* __restrict__ thr,
                                                        int N) {
    int c = threadIdx.x;
    float inv_n = 1.0f / (float)N;
    float mean  = sum[c] * inv_n;
    float var   = fmaxf(sumsq[c] * inv_n - mean * mean, 0.0f);
    thr[c] = mean - b_bn[c] * sqrtf(var + EPS_BN) / w_bn[c];
}

// Pack W[COUT, CIN] (f32, row-major) into f16 WMMA B fragments:
// offset = ((kt*32 + nt)*32 + lane)*16 + e
//   col n = nt*16 + (lane & 15);  k = kt*32 + (e&7) + (lane>>4)*8 + (e>>3)*16
__global__ __launch_bounds__(256) void pack_w_kernel(const float* __restrict__ W,
                                                     _Float16* __restrict__ Wp) {
    int idx = blockIdx.x * 256 + threadIdx.x;      // 0 .. 262143
    int e    = idx & 15;
    int lane = (idx >> 4) & 31;
    int nt   = (idx >> 9) & 31;
    int kt   = idx >> 14;
    int n = nt * 16 + (lane & 15);
    int k = kt * 32 + (e & 7) + (lane >> 4) * 8 + (e >> 3) * 16;
    Wp[idx] = (_Float16)W[(size_t)n * CIN + k];
}

// TDM: copy one contiguous 16KB (8192 f16) block from global to LDS.
__device__ __forceinline__ void tdm_copy_16k(unsigned lds_addr, const _Float16* src) {
    unsigned long long ga = (unsigned long long)(uintptr_t)src;
    u32x4 g0 = { 1u,
                 lds_addr,
                 (unsigned)ga,
                 (unsigned)((ga >> 32) & 0x01FFFFFFu) | 0x80000000u };
    i32x8 g1 = { 0x00010000,   // data_size=2B (bits 17:16 = 1)
                 0x20000000,   // tensor_dim0[15:0]=8192 in bits[63:48]
                 0x00010000,   // tensor_dim1[15:0]=1 in bits[95:80]
                 0x20000000,   // tile_dim0=8192 in bits[127:112]
                 1,            // tile_dim1=1, tile_dim2=0
                 8192,         // tensor_dim0_stride[31:0]
                 0, 0 };
    i32x4 z4 = { 0, 0, 0, 0 };
#if __clang_major__ >= 23
    i32x8 z8 = { 0, 0, 0, 0, 0, 0, 0, 0 };
    __builtin_amdgcn_tensor_load_to_lds(g0, g1, z4, z4, z8, 0);
#else
    __builtin_amdgcn_tensor_load_to_lds(g0, g1, z4, z4, 0);
#endif
}

// GEMM: out[n,o] = sum_c (x[n,c] > t[c]) * W[o,c] + bias[o]
// Block = 256 threads (8 waves). Block tile: M=128, N=256.
// TDM double-buffers B tiles in LDS; B fragments ping-pong between two live
// register sets so each WMMA overlaps the next fragment's LDS load; next
// K-step's x is prefetched before the WMMA chain.
__global__ __launch_bounds__(256) void spike_gemm_kernel(const float* __restrict__ x,
                                                         const float* __restrict__ thr,
                                                         const _Float16* __restrict__ Wp,
                                                         const float* __restrict__ bias,
                                                         float* __restrict__ out) {
    __shared__ __align__(32) _Float16 sB[2][16 * 32 * 16]; // 2 x 16KB B tiles
    __shared__ float sT[CIN];                              // thresholds, 2KB

    const int tid  = threadIdx.x;
    const int wave = tid >> 5;
    const int lane = tid & 31;
    const int g    = lane >> 4;            // lane half-group
    const int mBase = blockIdx.x * 128 + wave * 16;
    const int nBase = blockIdx.y * 256;
    const int mrow  = mBase + (lane & 15); // A-matrix row for this lane

    const _Float16* wpBase = Wp + (size_t)blockIdx.y * 16 * 512; // this block's N slice
    const unsigned lds0 = (unsigned)(uintptr_t)&sB[0][0];        // LDS aperture: low 32 bits

    for (int i = tid; i < CIN; i += 256) sT[i] = thr[i];

    if (wave == 0)                          // TDM ignores EXEC; one issue per wave
        tdm_copy_16k(lds0, wpBase);         // kt=0 -> buffer 0

    // Prefetch x for kt=0: A-layout K ranges [g*8, g*8+8) and [16+g*8, ...+8)
    const float* xrow0 = x + (size_t)mrow * CIN + g * 8;
    float4 xf0 = *(const float4*)(xrow0);
    float4 xf1 = *(const float4*)(xrow0 + 4);
    float4 xf2 = *(const float4*)(xrow0 + 16);
    float4 xf3 = *(const float4*)(xrow0 + 20);

    v8f acc[16];
#pragma unroll
    for (int nt = 0; nt < 16; ++nt) acc[nt] = (v8f){0.f,0.f,0.f,0.f,0.f,0.f,0.f,0.f};

    for (int kt = 0; kt < 16; ++kt) {      // K = 512 in steps of 32
        if (wave == 0) {
            if (kt < 15) {                 // prefetch next tile into other buffer
                tdm_copy_16k(lds0 + ((unsigned)((kt + 1) & 1)) * 16384u,
                             wpBase + (size_t)(kt + 1) * 16384);
                __builtin_amdgcn_s_wait_tensorcnt(1);  // TDM(kt) complete (in-order)
            } else {
                __builtin_amdgcn_s_wait_tensorcnt(0);
            }
        }
        __syncthreads();                   // buffer kt&1 ready (also covers sT on kt==0)

        // Build A fragment from prefetched x vs thresholds.
        const float* trow = sT + kt * 32 + g * 8;
        float4 tf0 = *(const float4*)(trow);
        float4 tf1 = *(const float4*)(trow + 4);
        float4 tf2 = *(const float4*)(trow + 16);
        float4 tf3 = *(const float4*)(trow + 20);
        const _Float16 ONE = (_Float16)1.0f, ZERO = (_Float16)0.0f;
        v16h a;
        a[0]  = (xf0.x > tf0.x) ? ONE : ZERO;
        a[1]  = (xf0.y > tf0.y) ? ONE : ZERO;
        a[2]  = (xf0.z > tf0.z) ? ONE : ZERO;
        a[3]  = (xf0.w > tf0.w) ? ONE : ZERO;
        a[4]  = (xf1.x > tf1.x) ? ONE : ZERO;
        a[5]  = (xf1.y > tf1.y) ? ONE : ZERO;
        a[6]  = (xf1.z > tf1.z) ? ONE : ZERO;
        a[7]  = (xf1.w > tf1.w) ? ONE : ZERO;
        a[8]  = (xf2.x > tf2.x) ? ONE : ZERO;
        a[9]  = (xf2.y > tf2.y) ? ONE : ZERO;
        a[10] = (xf2.z > tf2.z) ? ONE : ZERO;
        a[11] = (xf2.w > tf2.w) ? ONE : ZERO;
        a[12] = (xf3.x > tf3.x) ? ONE : ZERO;
        a[13] = (xf3.y > tf3.y) ? ONE : ZERO;
        a[14] = (xf3.z > tf3.z) ? ONE : ZERO;
        a[15] = (xf3.w > tf3.w) ? ONE : ZERO;

        // Prefetch x for kt+1 (in flight during the 16 WMMAs below).
        if (kt < 15) {
            const float* xn = xrow0 + (kt + 1) * 32;
            xf0 = *(const float4*)(xn);
            xf1 = *(const float4*)(xn + 4);
            xf2 = *(const float4*)(xn + 16);
            xf3 = *(const float4*)(xn + 20);
        }

        // WMMA chain: B ping-pongs between two register sets whose live
        // ranges overlap, so load(nt+1) issues before wmma(nt) completes.
        const _Float16* bLane = sB[kt & 1] + lane * 16;   // fragment nt at +nt*512
        v16h b0 = *(const v16h*)(bLane);
        v16h b1;
#pragma unroll
        for (int nt = 0; nt < 16; nt += 2) {
            b1 = *(const v16h*)(bLane + (nt + 1) * 512);
            acc[nt] = __builtin_amdgcn_wmma_f32_16x16x32_f16(
                false, a, false, b0, (short)0, acc[nt], false, false);
            if (nt + 2 < 16)
                b0 = *(const v16h*)(bLane + (nt + 2) * 512);
            acc[nt + 1] = __builtin_amdgcn_wmma_f32_16x16x32_f16(
                false, a, false, b1, (short)0, acc[nt + 1], false, false);
        }

        __syncthreads();                   // all reads done before this buffer is re-filled
    }

    // Epilogue: C/D layout -> lane%16 = N col, VGPR r -> M row = r + g*8.
#pragma unroll
    for (int nt = 0; nt < 16; ++nt) {
        int col = nBase + nt * 16 + (lane & 15);
        float bv = bias[col];
#pragma unroll
        for (int r = 0; r < 8; ++r) {
            int row = mBase + r + g * 8;
            out[(size_t)row * COUT + col] = acc[nt][r] + bv;
        }
    }
}

extern "C" void kernel_launch(void* const* d_in, const int* in_sizes, int n_in,
                              void* d_out, int out_size, void* d_ws, size_t ws_size,
                              hipStream_t stream) {
    const float* x    = (const float*)d_in[0];
    const float* w_bn = (const float*)d_in[1];
    const float* b_bn = (const float*)d_in[2];
    const float* w_fc = (const float*)d_in[3];
    const float* b_fc = (const float*)d_in[4];
    float* out = (float*)d_out;

    const int N = in_sizes[0] / CIN;       // 131072

    float*    ws_sum   = (float*)d_ws;
    float*    ws_sumsq = ws_sum + 512;
    float*    ws_thr   = ws_sum + 1024;
    _Float16* ws_wp    = (_Float16*)((char*)d_ws + 8192);

    init_stats_kernel<<<4, 256, 0, stream>>>(ws_sum);
    bn_stats_kernel<<<256, 512, 0, stream>>>(x, ws_sum, ws_sumsq, N);
    threshold_kernel<<<1, 512, 0, stream>>>(ws_sum, ws_sumsq, w_bn, b_bn, ws_thr, N);
    pack_w_kernel<<<(COUT * CIN) / 256, 256, 0, stream>>>(w_fc, ws_wp);

    dim3 grid(N / 128, COUT / 256);        // 1024 x 2 blocks
    spike_gemm_kernel<<<grid, 256, 0, stream>>>(x, ws_thr, ws_wp, b_fc, out);
}
